// HeteroGraphConvModel_48223892800090
// MI455X (gfx1250) — matrix operand order repaired and checked by reading, not compiled, not run on hardware
//
#include <hip/hip_runtime.h>
#include <hip/hip_bf16.h>
#include <cstdint>
#include <cstddef>

// ---------------------------------------------------------------------------
// CDNA5 / gfx1250 WMMA types (wave32; 16x16x32 bf16 -> f32 accum)
// ---------------------------------------------------------------------------
typedef __attribute__((ext_vector_type(16))) __bf16 v16bf;
typedef __attribute__((ext_vector_type(8)))  float  v8f;

union BF16Frag { v16bf v; uint4 q[2]; };

#define FLAG_LRELU 1
#define FLAG_ACCUM 2

// ---------------------------------------------------------------------------
// GEMM: C[rowtile, col0..col0+TN*16) (+)= lrelu( A(f32,MxK) @ W(bf16) + bias )
//
//  * Wt is the weight pre-converted to bf16 and TRANSPOSED: Wt[n*K + k].
//    Each lane's B fragment (column = wn*16+(lane&15), 16 contiguous K per
//    half-wave, same k order as the A fragment) is two global_load_b128 from L2.
//  * K-step 64: A tile = TM*16 rows x 64 K = TM*16*16 float4 slots -> exactly
//    tid-mapped (no guards/loops for TM=1); two WMMAs per barrier round.
//  * Register double-buffering: tile i+1 is loaded into regs while tile i's
//    WMMAs run; converted f32->bf16 and stored to LDS next iteration.
//  * M-tail by row clamping (C rows depend only on their own A row; clamped
//    rows are never stored) -> staging stays branch-free.
//  * LDS rows padded to 72 bf16 (144B = 36 banks) to de-conflict the 16
//    row-readers per half-wave.
// ---------------------------------------------------------------------------
template<int TM, int TN>
__global__ __launch_bounds__(256) void gemm_bf16_wmma_kernel(
    const float* __restrict__ A, int lda,
    const __bf16* __restrict__ Wt,       // N x K (transposed), N == TN*16
    const float* __restrict__ bias,      // length N, or nullptr
    float* __restrict__ C, int ldc, int col0,
    int M, int K, int flags)
{
    __shared__ __bf16 Asm[TM * 16][72];      // 64 K per row + 8 pad (bank skew)

    const int tid     = threadIdx.x;
    const int lane    = tid & 31;
    const int wave    = tid >> 5;
    const int wm      = wave / TN;
    const int wn      = wave % TN;
    const int rowbase = blockIdx.x * (TM * 16);
    const int mclamp  = M - 1;

    // staging slot -> (row, k-offset) ; slot s handles 4 consecutive f32
    int srow[TM], sko[TM];
    const float* sap[TM];
    #pragma unroll
    for (int i = 0; i < TM; ++i) {
        int s  = tid + i * 256;
        srow[i] = s >> 4;
        sko[i]  = (s & 15) * 4;
        int gr = rowbase + srow[i]; if (gr > mclamp) gr = mclamp;
        sap[i] = &A[(size_t)gr * lda + sko[i]];
    }

    // per-lane fragment addressing (k = 16*(lane>>4) + s for both A and B)
    const __bf16* Bp = Wt + (size_t)(wn * 16 + (lane & 15)) * K + (lane >> 4) * 16;
    const __bf16* Ap = &Asm[wm * 16 + (lane & 15)][(lane >> 4) * 16];

    v8f acc = {};
    float4 st[TM];

    // prologue: load first A tile into registers
    #pragma unroll
    for (int i = 0; i < TM; ++i) st[i] = *reinterpret_cast<const float4*>(sap[i]);

    for (int k0 = 0; k0 < K; k0 += 64) {
        // ---- convert staged regs -> LDS (2x v_cvt_pk_bf16_f32 + ds_store_b64)
        #pragma unroll
        for (int i = 0; i < TM; ++i) {
            union { __bf16 h[4]; uint2 u; } pk;
            pk.h[0] = (__bf16)st[i].x; pk.h[1] = (__bf16)st[i].y;
            pk.h[2] = (__bf16)st[i].z; pk.h[3] = (__bf16)st[i].w;
            *reinterpret_cast<uint2*>(&Asm[srow[i]][sko[i]]) = pk.u;
        }
        __syncthreads();

        // ---- start loading NEXT tile (overlaps with WMMAs below)
        if (k0 + 64 < K) {
            #pragma unroll
            for (int i = 0; i < TM; ++i)
                st[i] = *reinterpret_cast<const float4*>(sap[i] + (k0 + 64));
        }
        if (k0 + 128 < K)   // prefetch tile after that (global_prefetch_b8)
            __builtin_prefetch(sap[0] + (k0 + 128), 0, 1);

        // ---- two 16x16x32 WMMAs per barrier round
        BF16Frag af0, af1, bf0, bf1;
        af0.q[0] = *reinterpret_cast<const uint4*>(Ap);
        af0.q[1] = *reinterpret_cast<const uint4*>(Ap + 8);
        af1.q[0] = *reinterpret_cast<const uint4*>(Ap + 32);
        af1.q[1] = *reinterpret_cast<const uint4*>(Ap + 40);
        const __bf16* bp = Bp + k0;
        bf0.q[0] = *reinterpret_cast<const uint4*>(bp);
        bf0.q[1] = *reinterpret_cast<const uint4*>(bp + 8);
        bf1.q[0] = *reinterpret_cast<const uint4*>(bp + 32);
        bf1.q[1] = *reinterpret_cast<const uint4*>(bp + 40);

        acc = __builtin_amdgcn_wmma_f32_16x16x32_bf16(
            false, af0.v, false, bf0.v, (short)0, acc, false, false);
        acc = __builtin_amdgcn_wmma_f32_16x16x32_bf16(
            false, af1.v, false, bf1.v, (short)0, acc, false, false);

        __syncthreads();
    }

    // ---- epilogue: C layout = VGPR i -> M = i (+8 for hi half-wave), N = lane&15
    const int nl = wn * 16 + (lane & 15);
    const float bv = bias ? bias[nl] : 0.0f;
    #pragma unroll
    for (int i = 0; i < 8; ++i) {
        int gr = rowbase + wm * 16 + i + ((lane >> 4) ? 8 : 0);
        if (gr < M) {
            size_t ci = (size_t)gr * ldc + col0 + nl;
            float v = acc[i] + bv;
            if (flags & FLAG_ACCUM) v += C[ci];
            if (flags & FLAG_LRELU) v = (v > 0.0f) ? v : 0.01f * v;
            C[ci] = v;
        }
    }
}

// ---------------------------------------------------------------------------
// Helper kernels
// ---------------------------------------------------------------------------

// weight prep: f32 (K x N row-major) -> bf16 transposed (N x K): out[n*K+k]
__global__ void convert_transpose_kernel(const float* __restrict__ in,
                                         __bf16* __restrict__ out, int K, int N) {
    int idx = blockIdx.x * 256 + threadIdx.x;   // idx = n*K + k (coalesced writes)
    if (idx >= K * N) return;
    int n = idx / K;
    int k = idx - n * K;
    out[idx] = (__bf16)in[(size_t)k * N + n];
}

__global__ void zero4_kernel(float4* __restrict__ p, size_t n4) {
    size_t i = (size_t)blockIdx.x * 256 + threadIdx.x;
    if (i < n4) p[i] = float4{0.f, 0.f, 0.f, 0.f};
}

// user_feat columns 0..63: tiny K (num K=5, cat K=3) linears + lrelu
__global__ void encode_small_kernel(const float* __restrict__ ux,
                                    const float* __restrict__ Wn, const float* __restrict__ bn_,
                                    const float* __restrict__ Wc, const float* __restrict__ bc,
                                    float* __restrict__ uf, int M) {
    int idx = blockIdx.x * 256 + threadIdx.x;
    int row = idx >> 6;
    int j   = idx & 63;
    if (row >= M) return;
    const float* xr = ux + (size_t)row * 1544;
    float v;
    if (j < 32) {
        v = bn_[j];
        #pragma unroll
        for (int k = 0; k < 5; ++k) v += xr[k] * Wn[k * 32 + j];
    } else {
        int jj = j - 32;
        v = bc[jj];
        #pragma unroll
        for (int k = 0; k < 3; ++k) v += xr[5 + k] * Wc[k * 32 + jj];
    }
    v = (v > 0.0f) ? v : 0.01f * v;
    uf[(size_t)row * 128 + j] = v;
}

// segment_sum: 2 edges per block, 128 lanes per edge (dst tables live in 192MB L2)
__global__ void scatter_kernel(const float* __restrict__ srcf,
                               const int* __restrict__ edge, int E,
                               float* __restrict__ agg, float* __restrict__ cnt) {
    int e = blockIdx.x * 2 + (threadIdx.x >> 7);
    int f = threadIdx.x & 127;
    if (e >= E) return;
    int s = edge[e];
    int d = edge[E + e];
    atomicAdd(&agg[(size_t)d * 128 + f], srcf[(size_t)s * 128 + f]);
    if (f == 0) atomicAdd(&cnt[d], 1.0f);
}

__global__ void mean_kernel(float4* __restrict__ agg, const float* __restrict__ cnt, int M) {
    size_t i = (size_t)blockIdx.x * 256 + threadIdx.x;      // i indexes float4
    if (i >= (size_t)M * 32) return;
    int row = (int)(i >> 5);
    float inv = 1.0f / fmaxf(cnt[row], 1.0f);
    float4 a = agg[i];
    a.x *= inv; a.y *= inv; a.z *= inv; a.w *= inv;
    agg[i] = a;
}

__global__ void bn_stats_kernel(const float* __restrict__ x, int M,
                                float* __restrict__ sum, float* __restrict__ sq) {
    __shared__ float sh0[256];
    __shared__ float sh1[256];
    int col  = threadIdx.x & 127;
    int half = threadIdx.x >> 7;
    int base = blockIdx.x * 512;
    float s = 0.0f, s2 = 0.0f;
    for (int r = base + half; r < base + 512 && r < M; r += 2) {
        float v = x[(size_t)r * 128 + col];
        s += v; s2 += v * v;
    }
    sh0[threadIdx.x] = s;
    sh1[threadIdx.x] = s2;
    __syncthreads();
    if (half == 0) {
        atomicAdd(&sum[col], s  + sh0[threadIdx.x + 128]);
        atomicAdd(&sq[col],  s2 + sh1[threadIdx.x + 128]);
    }
}

__global__ void bn_apply_kernel(const float4* __restrict__ x, int M,
                                const float4* __restrict__ sum, const float4* __restrict__ sq,
                                const float4* __restrict__ g, const float4* __restrict__ b,
                                float4* __restrict__ out) {
    size_t i = (size_t)blockIdx.x * 256 + threadIdx.x;      // i indexes float4
    if (i >= (size_t)M * 32) return;
    int c4 = (int)(i & 31);                                  // float4-group within row
    float invM = 1.0f / (float)M;
    float4 s = sum[c4], q = sq[c4], gg = g[c4], bb = b[c4], v = x[i], o;
    float m, var;
    m = s.x * invM; var = q.x * invM - m * m; o.x = (v.x - m) * (gg.x * rsqrtf(var + 1e-5f)) + bb.x;
    m = s.y * invM; var = q.y * invM - m * m; o.y = (v.y - m) * (gg.y * rsqrtf(var + 1e-5f)) + bb.y;
    m = s.z * invM; var = q.z * invM - m * m; o.z = (v.z - m) * (gg.z * rsqrtf(var + 1e-5f)) + bb.z;
    m = s.w * invM; var = q.w * invM - m * m; o.w = (v.w - m) * (gg.w * rsqrtf(var + 1e-5f)) + bb.w;
    out[i] = o;
}

__global__ void concat_user_kernel(const float4* __restrict__ eu, const float4* __restrict__ uf,
                                   float4* __restrict__ out, int M) {
    size_t i = (size_t)blockIdx.x * 256 + threadIdx.x;      // out row = 256 f32 = 64 f4
    if (i >= (size_t)M * 64) return;
    int row = (int)(i >> 6);
    int c4  = (int)(i & 63);
    out[i] = (c4 < 32) ? eu[(size_t)row * 32 + c4]
                       : uf[(size_t)row * 32 + (c4 - 32)];
}

__global__ void concat_tweet_kernel(const float4* __restrict__ et, const float4* __restrict__ tx,
                                    float4* __restrict__ out, int M) {
    size_t i = (size_t)blockIdx.x * 256 + threadIdx.x;      // out row = 896 f32 = 224 f4
    if (i >= (size_t)M * 224) return;
    int row = (int)(i / 224);
    int c4  = (int)(i - (size_t)row * 224);
    out[i] = (c4 < 32) ? et[(size_t)row * 32 + c4]
                       : tx[(size_t)row * 192 + (c4 - 32)];
}

// ---------------------------------------------------------------------------
// Host orchestration
// ---------------------------------------------------------------------------
extern "C" void kernel_launch(void* const* d_in, const int* in_sizes, int n_in,
                              void* d_out, int out_size, void* d_ws, size_t ws_size,
                              hipStream_t stream)
{
    (void)in_sizes; (void)n_in; (void)out_size; (void)ws_size;

    const int NU = 100000, NT = 100000, E = 500000;

    // --- inputs (top-level dict insertion order, then params as sorted-key pytree) ---
    const float* user_x  = (const float*)d_in[0];
    const float* tweet_x = (const float*)d_in[1];
    const int*   e_follow = (const int*)d_in[2];
    const int*   e_friend = (const int*)d_in[3];
    const int*   e_post   = (const int*)d_in[4];
    const int*   e_rev    = (const int*)d_in[5];
    // params sorted keys: bn, cat, des, layers, lin_tweet, lin_user, num, twe
    const float* bn_g[2] = { (const float*)d_in[6], (const float*)d_in[8] };
    const float* bn_b[2] = { (const float*)d_in[7], (const float*)d_in[9] };
    const float* catW = (const float*)d_in[10]; const float* catB = (const float*)d_in[11];
    const float* desW = (const float*)d_in[12]; const float* desB = (const float*)d_in[13];
    // layers: d_in[14..37]: [l][follow,friend,post,rev_post][Wl,bl,Wr]
    auto LP = [&](int l, int et, int which) -> const float* {
        return (const float*)d_in[14 + l * 12 + et * 3 + which];
    };
    const float* ltW = (const float*)d_in[38]; const float* ltB = (const float*)d_in[39];
    const float* luW = (const float*)d_in[40]; const float* luB = (const float*)d_in[41];
    const float* numW = (const float*)d_in[42]; const float* numB = (const float*)d_in[43];
    const float* tweW = (const float*)d_in[44]; const float* tweB = (const float*)d_in[45];

    // --- workspace layout ---
    char* ws = (char*)d_ws;
    const size_t SZ = (size_t)NU * 128 * sizeof(float);   // 51.2 MB
    float* uf    = (float*)(ws + 0 * SZ);
    float* emb_u = (float*)(ws + 1 * SZ);
    float* emb_t = (float*)(ws + 2 * SZ);
    float* new_u = (float*)(ws + 3 * SZ);
    float* new_t = (float*)(ws + 4 * SZ);
    float* agg   = (float*)(ws + 5 * SZ);
    float* cnt   = (float*)(ws + 6 * SZ);
    float* bnsum = cnt + NU;
    float* bnsq  = bnsum + 128;
    __bf16* wb   = (__bf16*)(bnsq + 128);     // bf16 transposed-weight arena
    __bf16* Wdes = wb;                        // 32 x 768
    __bf16* Wtwe = wb + 24576;                // 32 x 768
    __bf16* Wlu  = wb + 49152;                // 128 x 128
    __bf16* Wlt  = wb + 65536;                // 128 x 768
    __bf16* Wlay = wb + 163840;               // 16 x (128*128): [l][et][Wl,Wr]
    auto WL = [&](int l, int et) { return Wlay + ((size_t)((l * 4 + et) * 2 + 0)) * 16384; };
    auto WR = [&](int l, int et) { return Wlay + ((size_t)((l * 4 + et) * 2 + 1)) * 16384; };

    // --- helpers ---
    auto cvtT = [&](const float* in, __bf16* out, int K, int N) {
        int n = K * N;
        convert_transpose_kernel<<<(n + 255) / 256, 256, 0, stream>>>(in, out, K, N);
    };
    auto zero = [&](float* p, size_t n) {   // n multiple of 4
        size_t n4 = n / 4;
        zero4_kernel<<<(unsigned)((n4 + 255) / 256), 256, 0, stream>>>((float4*)p, n4);
    };
    auto gemmN128 = [&](const float* A, int lda, const __bf16* W, const float* bias,
                        float* C, int ldc, int col0, int M, int K, int flags) {
        gemm_bf16_wmma_kernel<1, 8><<<(M + 15) / 16, 256, 0, stream>>>(
            A, lda, W, bias, C, ldc, col0, M, K, flags);
    };
    auto gemmN32 = [&](const float* A, int lda, const __bf16* W, const float* bias,
                       float* C, int ldc, int col0, int M, int K, int flags) {
        gemm_bf16_wmma_kernel<4, 2><<<(M + 63) / 64, 256, 0, stream>>>(
            A, lda, W, bias, C, ldc, col0, M, K, flags);
    };
    auto sage_mean = [&](const float* srcfeat, const int* edge) {
        zero(agg, (size_t)NU * 128);
        zero(cnt, NU);
        scatter_kernel<<<(E + 1) / 2, 256, 0, stream>>>(srcfeat, edge, E, agg, cnt);
        mean_kernel<<<(unsigned)(((size_t)NU * 32 + 255) / 256), 256, 0, stream>>>(
            (float4*)agg, cnt, NU);
    };

    // --- convert+transpose all GEMM weights to bf16 once per call ---
    cvtT(desW, Wdes, 768, 32);
    cvtT(tweW, Wtwe, 768, 32);
    cvtT(luW,  Wlu,  128, 128);
    cvtT(ltW,  Wlt,  768, 128);
    for (int l = 0; l < 2; ++l)
        for (int et = 0; et < 4; ++et) {
            cvtT(LP(l, et, 0), WL(l, et), 128, 128);
            cvtT(LP(l, et, 2), WR(l, et), 128, 128);
        }

    // --- feature encoders ---
    encode_small_kernel<<<(unsigned)(((size_t)NU * 64 + 255) / 256), 256, 0, stream>>>(
        user_x, numW, numB, catW, catB, uf, NU);
    gemmN32(user_x + 8,   1544, Wdes, desB, uf, 128, 64, NU, 768, FLAG_LRELU);  // des -> cols 64..95
    gemmN32(user_x + 776, 1544, Wtwe, tweB, uf, 128, 96, NU, 768, FLAG_LRELU);  // twe -> cols 96..127
    gemmN128(uf,      128, Wlu, luB, emb_u, 128, 0, NU, 128, FLAG_LRELU);
    gemmN128(tweet_x, 768, Wlt, ltB, emb_t, 128, 0, NT, 768, FLAG_LRELU);

    // --- 2 hetero-SAGE layers ---
    for (int l = 0; l < 2; ++l) {
        // follow (u->u)
        sage_mean(emb_u, e_follow);
        gemmN128(agg,   128, WL(l, 0), LP(l, 0, 1), new_u, 128, 0, NU, 128, 0);
        gemmN128(emb_u, 128, WR(l, 0), nullptr,     new_u, 128, 0, NU, 128, FLAG_ACCUM);
        // friend (u->u)
        sage_mean(emb_u, e_friend);
        gemmN128(agg,   128, WL(l, 1), LP(l, 1, 1), new_u, 128, 0, NU, 128, FLAG_ACCUM);
        gemmN128(emb_u, 128, WR(l, 1), nullptr,     new_u, 128, 0, NU, 128, FLAG_ACCUM);
        // rev_post (t->u)
        sage_mean(emb_t, e_rev);
        gemmN128(agg,   128, WL(l, 3), LP(l, 3, 1), new_u, 128, 0, NU, 128, FLAG_ACCUM);
        gemmN128(emb_u, 128, WR(l, 3), nullptr,     new_u, 128, 0, NU, 128, FLAG_ACCUM);
        // post (u->t)
        sage_mean(emb_u, e_post);
        gemmN128(agg,   128, WL(l, 2), LP(l, 2, 1), new_t, 128, 0, NT, 128, 0);
        gemmN128(emb_t, 128, WR(l, 2), nullptr,     new_t, 128, 0, NT, 128, FLAG_ACCUM);
        // batch-norm
        zero(bnsum, 128); zero(bnsq, 128);
        bn_stats_kernel<<<(NU + 511) / 512, 256, 0, stream>>>(new_u, NU, bnsum, bnsq);
        bn_apply_kernel<<<(unsigned)(((size_t)NU * 32 + 255) / 256), 256, 0, stream>>>(
            (const float4*)new_u, NU, (const float4*)bnsum, (const float4*)bnsq,
            (const float4*)bn_g[l], (const float4*)bn_b[l], (float4*)emb_u);
        zero(bnsum, 128); zero(bnsq, 128);
        bn_stats_kernel<<<(NT + 511) / 512, 256, 0, stream>>>(new_t, NT, bnsum, bnsq);
        bn_apply_kernel<<<(unsigned)(((size_t)NT * 32 + 255) / 256), 256, 0, stream>>>(
            (const float4*)new_t, NT, (const float4*)bnsum, (const float4*)bnsq,
            (const float4*)bn_g[l], (const float4*)bn_b[l], (float4*)emb_t);
    }

    // --- outputs: concat(emb_u, user_feat) ; concat(emb_t, tweet_x) ---
    float* out0 = (float*)d_out;
    float* out1 = out0 + (size_t)NU * 256;
    concat_user_kernel<<<(unsigned)(((size_t)NU * 64 + 255) / 256), 256, 0, stream>>>(
        (const float4*)emb_u, (const float4*)uf, (float4*)out0, NU);
    concat_tweet_kernel<<<(unsigned)(((size_t)NT * 224 + 255) / 256), 256, 0, stream>>>(
        (const float4*)emb_t, (const float4*)tweet_x, (float4*)out1, NT);
}
